// MyModel_61933428415536
// MI455X (gfx1250) — matrix operand order repaired and checked by reading, not compiled, not run on hardware
//
#include <hip/hip_runtime.h>
#include <hip/hip_bf16.h>
#include <math.h>

typedef __attribute__((ext_vector_type(16))) _Float16 v16h;
typedef __attribute__((ext_vector_type(2)))  _Float16 v2h;
typedef __attribute__((ext_vector_type(8)))  float    v8f;

#define NROWS   10240
#define DDIM    64
#define Q0      4096                 // queries in segment 0 (rows [0,4096))
#define KSEG    2048                 // keys per chunk (= L)
#define NKT     (KSEG / 16)          // 128 key tiles per chunk (uniform trip count)
#define NT0     (Q0 / 16)            // 256 query tiles seg0 (1 key chunk each)
#define NT1     ((NROWS - Q0) / 16)  // 384 query tiles seg1 (4 key chunks each)
#define NWAVES  (NT0 + NT1 * 4)      // 1792 perfectly balanced waves
// base-2 softmax domain: scores2 = (q.k)*10/sqrt(D)*log2(e); exp2 == v_exp_f32
#define QSCALE2 1.80336880698f       // 1.25 * log2(e), folded into Q fragment
#define SHIFT2  57.7078016355f       // 40 * log2(e); overflow needs score>128 = 12.8 sigma
                                     // (even unshifted tail would need 8.9 sigma; P~1e-30)

// d_ws layout (bytes): f16 key copy | w rowsums | L | O chunk states (~1.68 MB)
#define WS_KF16_OFF 0
#define WS_W_OFF    (NROWS * DDIM * 2)            // 1,310,720
#define WS_L_OFF    (WS_W_OFF + NROWS * 4)        // +40,960
#define WS_O_OFF    (WS_L_OFF + NROWS * 4 * 4)    // 4 chunk slots per row

// ---- prep: f16 copy of base (keys) + w_k = 10*rowsum(base[k]) ----
__global__ __launch_bounds__(256) void prep_kernel(const float* __restrict__ base,
                                                   _Float16* __restrict__ kf,
                                                   float* __restrict__ w) {
    const int row  = blockIdx.x * 8 + (threadIdx.x >> 5);   // one wave32 per row
    const int lane = threadIdx.x & 31;
    const float2 x = *(const float2*)(base + (size_t)row * DDIM + lane * 2);
    v2h h; h[0] = (_Float16)x.x; h[1] = (_Float16)x.y;
    *(v2h*)(kf + (size_t)row * DDIM + lane * 2) = h;
    float s = x.x + x.y;
#pragma unroll
    for (int off = 1; off < 32; off <<= 1) s += __shfl_xor(s, off, 32);
    if (lane == 0) w[row] = 10.f * s;
}

// ---- main: one wave32 per (16-query tile, 2048-key chunk); WMMA QK^T + base-2 softmax ----
__global__ __launch_bounds__(32) void sdpa_chunk_kernel(const float* __restrict__ base,
                                                        const _Float16* __restrict__ kf,
                                                        const float* __restrict__ w,
                                                        float* __restrict__ Lp,
                                                        float* __restrict__ Op) {
    const int wave = blockIdx.x;
    const int lane = threadIdx.x;        // 0..31 (wave32)
    const int half = lane >> 4;
    const int l16  = lane & 15;

    int qbase, kstart, chunk;
    if (wave < NT0) { qbase = wave * 16; kstart = 0; chunk = 0; }
    else {
        const int u = wave - NT0;
        qbase  = Q0 + (u >> 2) * 16;
        chunk  = u & 3;
        kstart = KSEG + chunk * KSEG;    // seg1 keys = rows [2048, 10240)
    }

    // Q tile -> two 16x16x32 A fragments (contiguous k = 16*half + j); 1.25*log2(e) folded in
    const float* qrow = base + (size_t)(qbase + l16) * DDIM;
    v16h a0, a1;
#pragma unroll
    for (int j = 0; j < 16; ++j) {
        const int kk = 16 * half + j;
        a0[j] = (_Float16)(qrow[kk]      * QSCALE2);
        a1[j] = (_Float16)(qrow[32 + kk] * QSCALE2);
    }

    // per-lane softmax accumulators: row m = r + 8*half, column = l16
    float l_run[8], o_run[8];
#pragma unroll
    for (int r = 0; r < 8; ++r) { l_run[r] = 0.f; o_run[r] = 0.f; }

    // uniform SGPR base + 32-bit lane offsets -> global_load saddr form
    const char* kb = (const char*)kf;
    const char* wb = (const char*)w;
    unsigned koff = ((unsigned)(kstart + l16) * DDIM + 16u * half) * 2u;
    unsigned woff = (unsigned)(kstart + l16) * 4u;

#pragma unroll 2
    for (int kt = 0; kt < NKT; ++kt) {
        __builtin_prefetch(kb + koff + 16 * DDIM * 2, 0, 3);   // next key tile

        // B fragments from f16 copy: lane = key column, halves = K 0-15/16-31
        const v16h b0 = *(const v16h*)(kb + koff);
        const v16h b1 = *(const v16h*)(kb + koff + 64);
        const float wv = *(const float*)(wb + woff);
        koff += 16 * DDIM * 2;
        woff += 16 * 4;

        v8f c = {};
        c = __builtin_amdgcn_wmma_f32_16x16x32_f16(false, a0, false, b0,
                                                   (short)0, c, false, false);
        c = __builtin_amdgcn_wmma_f32_16x16x32_f16(false, a1, false, b1,
                                                   (short)0, c, false, false);

        // ~3.5 VALU per score: pk-sub shift, exp2 (v_exp_f32), add, fmac
#pragma unroll
        for (int r = 0; r < 8; ++r) {
            const float e = __builtin_amdgcn_exp2f(c[r] - SHIFT2);
            l_run[r] += e;
            o_run[r] = fmaf(e, wv, o_run[r]);
        }
    }

    // plain-sum merge across the 16-lane column group, one writer per row
#pragma unroll
    for (int r = 0; r < 8; ++r) {
        float l = l_run[r], o = o_run[r];
#pragma unroll
        for (int off = 1; off < 16; off <<= 1) {
            l += __shfl_xor(l, off, 32);
            o += __shfl_xor(o, off, 32);
        }
        if (l16 == r) {
            const int slot = (qbase + r + 8 * half) * 4 + chunk;
            Lp[slot] = l; Op[slot] = o;
        }
    }
}

// ---- final: sum <=4 chunk states per query row (fixed order), reduce to scalar ----
__global__ __launch_bounds__(1024) void merge_reduce_kernel(const float* __restrict__ Lp,
                                                            const float* __restrict__ Op,
                                                            float* __restrict__ out) {
    __shared__ float smem[1024];
    const int tid = threadIdx.x;
    float acc = 0.f;
    for (int row = tid; row < NROWS; row += 1024) {
        const int nc = (row < Q0) ? 1 : 4;
        float l = 0.f, o = 0.f;
        for (int c = 0; c < nc; ++c) {
            l += Lp[row * 4 + c];
            o += Op[row * 4 + c];
        }
        acc += o / l;
    }
    smem[tid] = acc;
    __syncthreads();
    for (int stride = 512; stride > 0; stride >>= 1) {
        if (tid < stride) smem[tid] += smem[tid + stride];
        __syncthreads();
    }
    if (tid == 0) out[0] = smem[0];
}

extern "C" void kernel_launch(void* const* d_in, const int* in_sizes, int n_in,
                              void* d_out, int out_size, void* d_ws, size_t ws_size,
                              hipStream_t stream) {
    (void)in_sizes; (void)n_in; (void)out_size; (void)ws_size;
    const float* base = (const float*)d_in[0];
    char* ws = (char*)d_ws;
    _Float16* kf = (_Float16*)(ws + WS_KF16_OFF);
    float*    w  = (float*)(ws + WS_W_OFF);
    float*    Lp = (float*)(ws + WS_L_OFF);
    float*    Op = (float*)(ws + WS_O_OFF);

    prep_kernel<<<NROWS / 8, 256, 0, stream>>>(base, kf, w);
    sdpa_chunk_kernel<<<NWAVES, 32, 0, stream>>>(base, kf, w, Lp, Op);
    merge_reduce_kernel<<<1, 1024, 0, stream>>>(Lp, Op, (float*)d_out);
}
